// feature_embedding_5652176962177
// MI455X (gfx1250) — matrix operand order repaired and checked by reading, not compiled, not run on hardware
//
#include <hip/hip_runtime.h>
#include <hip/hip_bf16.h>
#include <math.h>

typedef __bf16 bf16;
typedef __attribute__((ext_vector_type(16))) bf16  v16bf;
typedef __attribute__((ext_vector_type(8)))  bf16  v8bf;
typedef __attribute__((ext_vector_type(8)))  float v8f;

#define DEST   64
#define NET    256
#define INPUT  448          // NET + 3*DEST
#define BATCH  4
#define NNODE  2048
#define ROWS   (BATCH*NNODE) // 8192

// ---------------- workspace layout (bytes) ----------------
#define SZ_S     (ROWS*128*2)        // struct bf16   : 2 MB
#define SZ_T     (ROWS*256*2)        // stat   bf16   : 4 MB
#define SZ_GATE  (ROWS*128*4)        // gate   f32    : 4 MB
#define SZ_H     (ROWS*128*2)        // H      bf16   : 2 MB (per dir)
#define OFF_S    0
#define OFF_T    (OFF_S + SZ_S)
#define OFF_GATE (OFF_T + SZ_T)
#define OFF_HF   (OFF_GATE + SZ_GATE)
#define OFF_HB   (OFF_HF + SZ_H)
#define OFF_WF   (OFF_HB + SZ_H)          // 128*64*2
#define OFF_WB   (OFF_WF + 128*64*2)
#define OFF_WE   (OFF_WB + 128*64*2)      // 256*128*2

#define LDS_STRIDE 136                    // 128 bf16 + 8 pad
#define LDS_STRIDE_W 72                   // 64 bf16 + 8 pad

static __device__ __forceinline__ v8f wmma_bf16(v16bf a, v16bf b, v8f c) {
    return __builtin_amdgcn_wmma_f32_16x16x32_bf16(false, a, false, b,
                                                   (short)0, c, false, false);
}

#define ASYNC_CHUNK16(LDSOFF, GPTR)                                           \
    asm volatile("global_load_async_to_lds_b128 %0, %1, off"                  \
                 :: "v"((unsigned)(LDSOFF)),                                  \
                    "v"((unsigned long long)(GPTR)) : "memory")

#define WAIT_ASYNC0() asm volatile("s_wait_asynccnt 0x0" ::: "memory")
#define WAIT_ASYNC2() asm volatile("s_wait_asynccnt 0x2" ::: "memory")

// ---------------------------------------------------------------------------
// prep: fp32 -> bf16 conversion of struct slice, stat slice, weight matrices
// ---------------------------------------------------------------------------
__global__ void prep_kernel(const float* __restrict__ nf,
                            const float* __restrict__ Wf,
                            const float* __restrict__ Wb,
                            const float* __restrict__ Wemb,
                            bf16* __restrict__ Sb, bf16* __restrict__ Tb,
                            bf16* __restrict__ Wfb, bf16* __restrict__ Wbb,
                            bf16* __restrict__ Web) {
    const int nS  = ROWS * 128;
    const int nT  = ROWS * 256;
    const int nW  = 128 * 64;
    const int nWe = 256 * 128;
    const int total = nS + nT + 2 * nW + nWe;
    int stride = gridDim.x * blockDim.x;
    for (int i = blockIdx.x * blockDim.x + threadIdx.x; i < total; i += stride) {
        if (i < nS) {
            int row = i >> 7, f = i & 127;
            Sb[i] = (bf16)nf[(size_t)row * INPUT + DEST + f];
        } else if (i < nS + nT) {
            int j = i - nS;
            int row = j >> 8, f = j & 255;
            Tb[j] = (bf16)nf[(size_t)row * INPUT + 3 * DEST + f];
        } else if (i < nS + nT + nW) {
            int j = i - nS - nT;
            Wfb[j] = (bf16)Wf[j];
        } else if (i < nS + nT + 2 * nW) {
            int j = i - nS - nT - nW;
            Wbb[j] = (bf16)Wb[j];
        } else {
            int j = i - nS - nT - 2 * nW;
            Web[j] = (bf16)Wemb[j];
        }
    }
}

// ---------------------------------------------------------------------------
// gate = sigmoid(stat @ W_emb + b_emb)   M=8192  K=256  N=128
// Entire W_emb (256x128 bf16) staged in LDS once per block.
// ---------------------------------------------------------------------------
__global__ __launch_bounds__(256) void gate_kernel(const bf16* __restrict__ Tb,
                                                   const bf16* __restrict__ Web,
                                                   const float* __restrict__ bemb,
                                                   float* __restrict__ gate) {
    __shared__ bf16 wlds[256 * LDS_STRIDE];
    const int t = threadIdx.x;
#pragma unroll
    for (int j = 0; j < 16; ++j) {
        int q = t + 256 * j;
        int row = q >> 4, cc = q & 15;           // 16 x 16B chunks per 256B row
        unsigned loff =
            (unsigned)(unsigned long long)(const void*)&wlds[row * LDS_STRIDE + cc * 8];
        const bf16* g = Web + (size_t)row * 128 + cc * 8;
        ASYNC_CHUNK16(loff, g);
    }
    WAIT_ASYNC0();
    __syncthreads();

    const int w = t >> 5;
    const int lane = t & 31;
    const int half = lane >> 4;
    const int lrow = lane & 15;
    const int m0 = blockIdx.x * 128 + w * 16;
    const int arow = m0 + lrow;
    const int koff = half ? 8 : 0;

    v8f acc[8];
#pragma unroll
    for (int n = 0; n < 8; ++n)
#pragma unroll
        for (int r = 0; r < 8; ++r) acc[n][r] = 0.0f;

    for (int k0 = 0; k0 < 256; k0 += 32) {
        v8bf alo = *(const v8bf*)(Tb + (size_t)arow * 256 + k0 + koff);
        v8bf ahi = *(const v8bf*)(Tb + (size_t)arow * 256 + k0 + 16 + koff);
        v16bf a;
#pragma unroll
        for (int i = 0; i < 8; ++i) { a[i] = alo[i]; a[8 + i] = ahi[i]; }
        const bf16* Brow = &wlds[(k0 + koff) * LDS_STRIDE + lrow];
#pragma unroll
        for (int n = 0; n < 8; ++n) {
            v16bf bv;
#pragma unroll
            for (int j = 0; j < 8; ++j) { bv[j] = Brow[j * LDS_STRIDE + n * 16]; bv[8 + j] = Brow[(16 + j) * LDS_STRIDE + n * 16]; }
            acc[n] = wmma_bf16(a, bv, acc[n]);
        }
    }
#pragma unroll
    for (int n = 0; n < 8; ++n) {
        int col = n * 16 + lrow;
        float bias = bemb[col];
#pragma unroll
        for (int r = 0; r < 8; ++r) {
            int row = m0 + r + 8 * half;
            float x = acc[n][r] + bias;
            gate[(size_t)row * 128 + col] = 1.0f / (1.0f + __expf(-x));
        }
    }
}

// ---------------------------------------------------------------------------
// H = adj @ struct   per (batch, dir):  (2048x2048) @ (2048x128) -> bf16
// grid: x = 16 row-tiles of 128, z = 8 (batch*2 + dir)
// block = 8 waves as 4(M) x 2(N); wave tile = 32 rows x 64 cols.
// B tile (32 x 128 bf16) staged in LDS, double-buffered via async-to-LDS.
// ---------------------------------------------------------------------------
__global__ __launch_bounds__(256) void adj_gemm_kernel(const float* __restrict__ adjF,
                                                       const float* __restrict__ adjB,
                                                       const bf16* __restrict__ Sb,
                                                       bf16* __restrict__ Hf,
                                                       bf16* __restrict__ Hb) {
    __shared__ bf16 smem[2][32 * LDS_STRIDE];

    const int z = blockIdx.z;
    const int b = z >> 1, dir = z & 1;
    const float* A = (dir ? adjB : adjF) + (size_t)b * NNODE * NNODE;
    const bf16*  S = Sb + (size_t)b * NNODE * 128;
    bf16*        H = (dir ? Hb : Hf) + (size_t)b * NNODE * 128;

    const int t = threadIdx.x;
    const int w = t >> 5, lane = t & 31;
    const int half = lane >> 4, lrow = lane & 15;
    const int rw = w & 3, cw = w >> 2;
    const int m0 = blockIdx.x * 128 + rw * 32;    // batch-local row base of wave tile
    const int koff = half ? 8 : 0;
    const float* Ar0 = A + (size_t)(m0 + lrow) * NNODE;
    const float* Ar1 = A + (size_t)(m0 + 16 + lrow) * NNODE;

    // cooperative staging map: 512 x 16B chunks per 32x128 tile, 2 per thread
    const int r0 = t >> 4, c0 = t & 15;           // rows 0..15
    const int r1 = r0 + 16;                       // rows 16..31
    unsigned lo[2][2];
#pragma unroll
    for (int bu = 0; bu < 2; ++bu) {
        lo[bu][0] = (unsigned)(unsigned long long)(const void*)&smem[bu][r0 * LDS_STRIDE + c0 * 8];
        lo[bu][1] = (unsigned)(unsigned long long)(const void*)&smem[bu][r1 * LDS_STRIDE + c0 * 8];
    }

    v8f acc[2][4];
#pragma unroll
    for (int mt = 0; mt < 2; ++mt)
#pragma unroll
        for (int n = 0; n < 4; ++n)
#pragma unroll
            for (int r = 0; r < 8; ++r) acc[mt][n][r] = 0.0f;

#define ISSUE_TILE(K0, BU)                                                    \
    do {                                                                      \
        const bf16* g0 = S + (size_t)((K0) + r0) * 128 + c0 * 8;              \
        const bf16* g1 = S + (size_t)((K0) + r1) * 128 + c0 * 8;              \
        asm volatile("global_load_async_to_lds_b128 %0, %2, off\n\t"          \
                     "global_load_async_to_lds_b128 %1, %3, off"              \
                     :: "v"(lo[BU][0]), "v"(lo[BU][1]),                       \
                        "v"((unsigned long long)g0),                          \
                        "v"((unsigned long long)g1)                           \
                     : "memory");                                             \
    } while (0)

    ISSUE_TILE(0, 0);
    const int KSTEPS = NNODE / 32;                // 64
    for (int i = 0; i < KSTEPS; ++i) {
        const int k0 = i * 32;
        if (i + 1 < KSTEPS) {
            ISSUE_TILE(k0 + 32, (i + 1) & 1);
            WAIT_ASYNC2();                        // tile i landed
        } else {
            WAIT_ASYNC0();
        }
        __syncthreads();

        __builtin_prefetch(Ar0 + k0 + 1024, 0, 0);
        __builtin_prefetch(Ar1 + k0 + 1024, 0, 0);
        v8f f0 = *(const v8f*)(Ar0 + k0 + koff);
        v8f f1 = *(const v8f*)(Ar0 + k0 + 16 + koff);
        v8f f2 = *(const v8f*)(Ar1 + k0 + koff);
        v8f f3 = *(const v8f*)(Ar1 + k0 + 16 + koff);
        v16bf a0, a1;
#pragma unroll
        for (int j = 0; j < 8; ++j) {
            a0[j] = (bf16)f0[j]; a0[8 + j] = (bf16)f1[j];
            a1[j] = (bf16)f2[j]; a1[8 + j] = (bf16)f3[j];
        }

        const bf16* brow = &smem[i & 1][koff * LDS_STRIDE + cw * 64 + lrow];
#pragma unroll
        for (int n = 0; n < 4; ++n) {
            v16bf bv;
#pragma unroll
            for (int j = 0; j < 8; ++j) { bv[j] = brow[j * LDS_STRIDE + n * 16]; bv[8 + j] = brow[(16 + j) * LDS_STRIDE + n * 16]; }
            acc[0][n] = wmma_bf16(a0, bv, acc[0][n]);
            acc[1][n] = wmma_bf16(a1, bv, acc[1][n]);
        }
        __syncthreads();
    }
#undef ISSUE_TILE

#pragma unroll
    for (int mt = 0; mt < 2; ++mt)
#pragma unroll
        for (int n = 0; n < 4; ++n) {
            int col = cw * 64 + n * 16 + lrow;
#pragma unroll
            for (int r = 0; r < 8; ++r) {
                int row = m0 + mt * 16 + r + 8 * half;
                H[(size_t)row * 128 + col] = (bf16)acc[mt][n][r];
            }
        }
}

// ---------------------------------------------------------------------------
// out[:, dir*64 + c] = dyn[:, c] * (H_dir @ W_dir + b_dir)[:, c] * gate[:, dir*64 + c]
// W (128x64 bf16): only cols 0..31 of each row staged in LDS once per block.
// ---------------------------------------------------------------------------
__global__ __launch_bounds__(256) void emb_kernel(const bf16* __restrict__ Hf,
                                                  const bf16* __restrict__ Hb,
                                                  const bf16* __restrict__ Wfb,
                                                  const bf16* __restrict__ Wbb,
                                                  const float* __restrict__ bfv,
                                                  const float* __restrict__ bbv,
                                                  const float* __restrict__ nf,
                                                  const float* __restrict__ gate,
                                                  float* __restrict__ out) {
    __shared__ bf16 wlds[128 * LDS_STRIDE_W];
    const int dir = blockIdx.z;
    const bf16*  H    = dir ? Hb : Hf;
    const bf16*  W    = dir ? Wbb : Wfb;
    const float* bias = dir ? bbv : bfv;

    const int t = threadIdx.x;
#pragma unroll
    for (int j = 0; j < 2; ++j) {
        int q = t + 256 * j;
        int row = q >> 2, cc = q & 3;             // 4 x 16B chunks: first 64B of each 128B row
        unsigned loff =
            (unsigned)(unsigned long long)(const void*)&wlds[row * LDS_STRIDE_W + cc * 8];
        const bf16* g = W + (size_t)row * 64 + cc * 8;
        ASYNC_CHUNK16(loff, g);
    }
    WAIT_ASYNC0();
    __syncthreads();

    const int w = t >> 5;
    const int lane = t & 31;
    const int half = lane >> 4;
    const int lrow = lane & 15;
    const int m0 = blockIdx.x * 128 + w * 16;     // global row base (0..8191)
    const int arow = m0 + lrow;
    const int koff = half ? 8 : 0;

    v8f acc[4];
#pragma unroll
    for (int n = 0; n < 4; ++n)
#pragma unroll
        for (int r = 0; r < 8; ++r) acc[n][r] = 0.0f;

#pragma unroll
    for (int k0 = 0; k0 < 128; k0 += 32) {
        v8bf alo = *(const v8bf*)(H + (size_t)arow * 128 + k0 + koff);
        v8bf ahi = *(const v8bf*)(H + (size_t)arow * 128 + k0 + 16 + koff);
        v16bf a;
#pragma unroll
        for (int i = 0; i < 8; ++i) { a[i] = alo[i]; a[8 + i] = ahi[i]; }
        const bf16* Brow = &wlds[(k0 + koff) * LDS_STRIDE_W + lrow];
#pragma unroll
        for (int n = 0; n < 4; ++n) {
            v16bf bv;
#pragma unroll
            for (int j = 0; j < 8; ++j) { bv[j] = Brow[j * LDS_STRIDE_W + n * 16]; bv[8 + j] = Brow[(16 + j) * LDS_STRIDE_W + n * 16]; }
            acc[n] = wmma_bf16(a, bv, acc[n]);
        }
    }
#pragma unroll
    for (int n = 0; n < 4; ++n) {
        int col = n * 16 + lrow;              // 0..63
        float bi = bias[col];
#pragma unroll
        for (int r = 0; r < 8; ++r) {
            int row = m0 + r + 8 * half;
            float h   = acc[n][r] + bi;
            float dyn = nf[(size_t)row * INPUT + col];
            float g   = gate[(size_t)row * 128 + dir * 64 + col];
            out[(size_t)row * 128 + dir * 64 + col] = dyn * h * g;
        }
    }
}

// ---------------------------------------------------------------------------
extern "C" void kernel_launch(void* const* d_in, const int* in_sizes, int n_in,
                              void* d_out, int out_size, void* d_ws, size_t ws_size,
                              hipStream_t stream) {
    (void)in_sizes; (void)n_in; (void)out_size; (void)ws_size;
    const float* nf   = (const float*)d_in[0];
    const float* adjF = (const float*)d_in[1];
    const float* adjB = (const float*)d_in[2];
    const float* Wf   = (const float*)d_in[3];
    const float* bfv  = (const float*)d_in[4];
    const float* Wb   = (const float*)d_in[5];
    const float* bbv  = (const float*)d_in[6];
    const float* Wemb = (const float*)d_in[7];
    const float* bemb = (const float*)d_in[8];
    float* out = (float*)d_out;

    char* ws = (char*)d_ws;
    bf16*  Sb   = (bf16*)(ws + OFF_S);
    bf16*  Tb   = (bf16*)(ws + OFF_T);
    float* gate = (float*)(ws + OFF_GATE);
    bf16*  Hf   = (bf16*)(ws + OFF_HF);
    bf16*  Hb   = (bf16*)(ws + OFF_HB);
    bf16*  Wfb  = (bf16*)(ws + OFF_WF);
    bf16*  Wbb  = (bf16*)(ws + OFF_WB);
    bf16*  Web  = (bf16*)(ws + OFF_WE);

    prep_kernel<<<dim3(2048), dim3(256), 0, stream>>>(nf, Wf, Wb, Wemb,
                                                      Sb, Tb, Wfb, Wbb, Web);
    gate_kernel<<<dim3(ROWS / 128), dim3(256), 0, stream>>>(Tb, Web, bemb, gate);
    adj_gemm_kernel<<<dim3(NNODE / 128, 1, BATCH * 2), dim3(256), 0, stream>>>(
        adjF, adjB, Sb, Hf, Hb);
    emb_kernel<<<dim3(ROWS / 128, 1, 2), dim3(256), 0, stream>>>(
        Hf, Hb, Wfb, Wbb, bfv, bbv, nf, gate, out);
}
